// Masker_8916352106695
// MI455X (gfx1250) — matrix operand order repaired and checked by reading, not compiled, not run on hardware
//
#include <hip/hip_runtime.h>
#include <hip/hip_bf16.h>

typedef __bf16 bf16_t;
typedef __attribute__((ext_vector_type(16))) __bf16 v16bf;
typedef __attribute__((ext_vector_type(8)))  float  v8f;

#define AST 36   // f32 elements per A-tile row (32 + 4 pad) -> 144B, 16B aligned
#define BST 36   // same for B tile; 36k mod 64 distinct for k=0..15 -> no bank conflicts

// Issue one per-lane 16B async global->LDS copy (ASYNCcnt-tracked DMA).
__device__ __forceinline__ void async_ld16(unsigned lds_addr, const float* g) {
  asm volatile("global_load_async_to_lds_b128 %0, %1, off"
               :: "v"(lds_addr), "v"((unsigned long long)(uintptr_t)g)
               : "memory");
}

// ---------------------------------------------------------------------------
// Fused GEMM (64 x N = 64 x K * K x N) + BatchNorm(axis=0) [+ReLU].
// One block owns all 64 batch rows of a 32-column stripe -> BN stats are a
// block-local LDS reduction. Biases skipped: they cancel under BatchNorm.
// 256 threads = 8 waves; wave w -> C tile rows 16*(w&3), cols 16*(w>>2).
// Double-buffered LDS fed by async copies; WMMA of chunk i overlaps the DMA
// of chunk i+1 (s_wait_asynccnt + barrier for cross-wave visibility).
// ---------------------------------------------------------------------------
__global__ __launch_bounds__(256)
void gemm64_bn_kernel(const float* __restrict__ A, const float* __restrict__ Bm,
                      const float* __restrict__ gamma, const float* __restrict__ beta,
                      float* __restrict__ Out, int Kd, int N, int affine_relu)
{
  __shared__ float sA[2][64 * AST];    // A tile 64x32 f32, row-major (K contiguous)
  __shared__ float sB[2][32 * BST];    // B tile 32x32 f32, row-major (N contiguous)
  __shared__ float redS[32][8];
  __shared__ float redQ[32][8];
  __shared__ float colA[32];
  __shared__ float colD[32];

  const int tid  = threadIdx.x;
  const int w    = tid >> 5;           // wave id 0..7
  const int ln   = tid & 31;
  const int r    = w & 3;              // row tile (16 rows each)
  const int c    = w >> 2;             // col tile (16 cols each)
  const int half = ln >> 4;            // lane group 0/1
  const int m15  = ln & 15;
  const int nb   = blockIdx.x * 32;    // global column base

  v8f acc = {};

  // staging thread mapping (16B chunks per lane)
  const int a_row = tid >> 2;          // 0..63
  const int a_kq  = (tid & 3) * 8;     // 0,8,16,24 (two float4 chunks)
  const int b_kr  = tid >> 3;          // 0..31
  const int b_nq  = (tid & 7) * 4;     // 0,4,...,28

  const int nchunks = Kd >> 5;

#define ISSUE_CHUNK(KC, BUF)                                                  \
  {                                                                           \
    const float* ga = A + (size_t)a_row * Kd + (KC) + a_kq;                   \
    unsigned la = (unsigned)(uintptr_t)&sA[(BUF)][a_row * AST + a_kq];        \
    async_ld16(la, ga);                                                       \
    async_ld16(la + 16u, ga + 4);                                             \
    const float* gb = Bm + (size_t)((KC) + b_kr) * N + nb + b_nq;             \
    unsigned lb = (unsigned)(uintptr_t)&sB[(BUF)][b_kr * BST + b_nq];         \
    async_ld16(lb, gb);                                                       \
  }

  ISSUE_CHUNK(0, 0)

  for (int ci = 0; ci < nchunks; ++ci) {
    const int buf = ci & 1;
    if (ci + 1 < nchunks) {
      ISSUE_CHUNK((ci + 1) << 5, buf ^ 1)      // prefetch next chunk's DMA
      if (ci + 2 < nchunks)                    // warm L2 two chunks ahead
        __builtin_prefetch(Bm + (size_t)(((ci + 2) << 5) + b_kr) * N + nb + b_nq, 0, 1);
      asm volatile("s_wait_asynccnt 0x3" ::: "memory");  // chunk ci landed (this wave)
    } else {
      asm volatile("s_wait_asynccnt 0x0" ::: "memory");
    }
    __syncthreads();                           // all waves' copies visible

    // ---- build bf16 fragments (convert at read time) ----
    v16bf af, bfr;
    {
      // A: lane<16 holds row m, K {0..7,16..23}; lane>=16 holds K {8..15,24..31}
      const float* ar = &sA[buf][(r * 16 + m15) * AST + half * 8];
      float4 a0 = *(const float4*)(ar);
      float4 a1 = *(const float4*)(ar + 4);
      float4 a2 = *(const float4*)(ar + 16);
      float4 a3 = *(const float4*)(ar + 20);
      af[0]=(bf16_t)a0.x; af[1]=(bf16_t)a0.y; af[2]=(bf16_t)a0.z; af[3]=(bf16_t)a0.w;
      af[4]=(bf16_t)a1.x; af[5]=(bf16_t)a1.y; af[6]=(bf16_t)a1.z; af[7]=(bf16_t)a1.w;
      af[8]=(bf16_t)a2.x; af[9]=(bf16_t)a2.y; af[10]=(bf16_t)a2.z; af[11]=(bf16_t)a2.w;
      af[12]=(bf16_t)a3.x; af[13]=(bf16_t)a3.y; af[14]=(bf16_t)a3.z; af[15]=(bf16_t)a3.w;
      // B: lane<16 holds col n, K {0..7,16..23}; lane>=16 holds K {8..15,24..31}
      // (column gather from row-major tile; stride 36 words is conflict-free)
      const float* bcol = &sB[buf][(half * 8) * BST + c * 16 + m15];
      #pragma unroll
      for (int k = 0; k < 16; ++k) bfr[k] = (bf16_t)bcol[(k + (k & 8)) * BST];
    }

    acc = __builtin_amdgcn_wmma_f32_16x16x32_bf16(false, af, false, bfr,
                                                  (short)0, acc, false, false);
    __syncthreads();                           // reads done before buffer reuse
  }
#undef ISSUE_CHUNK

  // ---- BatchNorm epilogue: stats over the 64 batch rows ----
  float s = 0.f, q = 0.f;
  #pragma unroll
  for (int j = 0; j < 8; ++j) { s += acc[j]; q += acc[j] * acc[j]; }
  const int col = c * 16 + m15;                // 0..31 within block
  redS[col][r * 2 + half] = s;
  redQ[col][r * 2 + half] = q;
  __syncthreads();
  if (tid < 32) {
    float S = 0.f, Q = 0.f;
    #pragma unroll
    for (int p = 0; p < 8; ++p) { S += redS[tid][p]; Q += redQ[tid][p]; }
    float mean = S * (1.0f / 64.0f);
    float var  = Q * (1.0f / 64.0f) - mean * mean;   // biased, matches reference
    float g  = affine_relu ? gamma[nb + tid] : 1.0f;
    float be = affine_relu ? beta [nb + tid] : 0.0f;
    float a  = g * rsqrtf(var + 1e-5f);
    colA[tid] = a;
    colD[tid] = be - mean * a;
  }
  __syncthreads();
  const float a = colA[col], d = colD[col];
  const int gcol = nb + col;
  #pragma unroll
  for (int j = 0; j < 8; ++j) {
    int row = r * 16 + half * 8 + j;           // C-matrix VGPR->row mapping
    float y = acc[j] * a + d;
    if (affine_relu) y = fmaxf(y, 0.0f);
    Out[(size_t)row * N + gcol] = y;
  }
}

// ---------------------------------------------------------------------------
// Gumbel-softmax loop: K=1024 sequential iterations over 64x2048.
// One block per row; mask/z live in registers; two block reductions per iter.
// ---------------------------------------------------------------------------
__device__ __forceinline__ unsigned rotl32(unsigned x, int n) {
  return (x << n) | (x >> (32 - n));
}

__device__ __forceinline__ unsigned threefry2x32(unsigned k0, unsigned k1,
                                                 unsigned c0, unsigned c1) {
  unsigned ks2 = k0 ^ k1 ^ 0x1BD11BDAu;
  unsigned x0 = c0 + k0, x1 = c1 + k1;
#define TF_RND(rot) { x0 += x1; x1 = rotl32(x1, rot); x1 ^= x0; }
  TF_RND(13) TF_RND(15) TF_RND(26) TF_RND(6)
  x0 += k1;  x1 += ks2 + 1u;
  TF_RND(17) TF_RND(29) TF_RND(16) TF_RND(24)
  x0 += ks2; x1 += k0 + 2u;
  TF_RND(13) TF_RND(15) TF_RND(26) TF_RND(6)
  x0 += k0;  x1 += k1 + 3u;
  TF_RND(17) TF_RND(29) TF_RND(16) TF_RND(24)
  x0 += k1;  x1 += ks2 + 4u;
  TF_RND(13) TF_RND(15) TF_RND(26) TF_RND(6)
  x0 += ks2; x1 += k0 + 5u;
#undef TF_RND
  return x0;
}

__global__ __launch_bounds__(256)
void gumbel_kernel(const float* __restrict__ mask0, float* __restrict__ zout)
{
  __shared__ float wred[8];
  const int row = blockIdx.x;
  const int tid = threadIdx.x;
  const int wv  = tid >> 5, lane = tid & 31;

  float m[8], z[8];
  #pragma unroll
  for (int k = 0; k < 8; ++k) {
    m[k] = mask0[row * 2048 + tid + k * 256];
    z[k] = 0.0f;
  }

  for (int it = 0; it < 1024; ++it) {
    float t[8];
    float lmax = -3.4e38f;
    #pragma unroll
    for (int k = 0; k < 8; ++k) {
      unsigned idx  = (unsigned)(row * 2048 + tid + k * 256);
      unsigned bits = threefry2x32(42u, (unsigned)it, idx, 0x9E3779B9u);
      float u = (float)(bits >> 8) * 5.9604645e-08f + 2.9802322e-08f; // (0,1)
      float g = -__logf(-__logf(u));
      t[k] = (m[k] + g) * 2.0f;          // /TAU, TAU = 0.5
      lmax = fmaxf(lmax, t[k]);
    }
    // block max
    #pragma unroll
    for (int o = 16; o > 0; o >>= 1) lmax = fmaxf(lmax, __shfl_xor(lmax, o, 32));
    if (lane == 0) wred[wv] = lmax;
    __syncthreads();
    float bmax = wred[0];
    #pragma unroll
    for (int i = 1; i < 8; ++i) bmax = fmaxf(bmax, wred[i]);
    __syncthreads();
    // exp + block sum
    float lsum = 0.0f;
    #pragma unroll
    for (int k = 0; k < 8; ++k) { t[k] = __expf(t[k] - bmax); lsum += t[k]; }
    #pragma unroll
    for (int o = 16; o > 0; o >>= 1) lsum += __shfl_xor(lsum, o, 32);
    if (lane == 0) wred[wv] = lsum;
    __syncthreads();
    float bsum = 0.0f;
    #pragma unroll
    for (int i = 0; i < 8; ++i) bsum += wred[i];
    __syncthreads();
    float inv = 1.0f / bsum;
    #pragma unroll
    for (int k = 0; k < 8; ++k) { m[k] = t[k] * inv; z[k] = fmaxf(z[k], m[k]); }
  }

  #pragma unroll
  for (int k = 0; k < 8; ++k) zout[row * 2048 + tid + k * 256] = z[k];
}

// ---------------------------------------------------------------------------
extern "C" void kernel_launch(void* const* d_in, const int* in_sizes, int n_in,
                              void* d_out, int out_size, void* d_ws, size_t ws_size,
                              hipStream_t stream)
{
  (void)in_sizes; (void)n_in; (void)out_size; (void)ws_size;
  const float* f   = (const float*)d_in[0];
  const float* W1  = (const float*)d_in[1];
  // d_in[2] = b1: cancels under BatchNorm, unused
  const float* g1  = (const float*)d_in[3];
  const float* be1 = (const float*)d_in[4];
  const float* W2  = (const float*)d_in[5];
  // d_in[6] = b2: unused
  const float* g2  = (const float*)d_in[7];
  const float* be2 = (const float*)d_in[8];
  const float* W3  = (const float*)d_in[9];
  // d_in[10] = b3: unused

  float* h1 = (float*)d_ws;            // 64 x 8192
  float* h2 = h1 + 64 * 8192;          // 64 x 8192
  float* m0 = h2 + 64 * 8192;          // 64 x 2048

  gemm64_bn_kernel<<<8192 / 32, 256, 0, stream>>>(f,  W1, g1, be1, h1, 2048, 8192, 1);
  gemm64_bn_kernel<<<8192 / 32, 256, 0, stream>>>(h1, W2, g2, be2, h2, 8192, 8192, 1);
  gemm64_bn_kernel<<<2048 / 32, 256, 0, stream>>>(h2, W3, nullptr, nullptr, m0, 8192, 2048, 0);
  gumbel_kernel<<<64, 256, 0, stream>>>(m0, (float*)d_out);
}
